// DeepFusedGPTTransformerLayer_77850577207454
// MI455X (gfx1250) — matrix operand order repaired and checked by reading, not compiled, not run on hardware
//
#include <hip/hip_runtime.h>
#include <hip/hip_bf16.h>

#define B_   2
#define S_   2048
#define H_   2048
#define NH_  16
#define D_   128
#define FF_  (4*H_)
#define MTOT (B_*S_)   // 4096 rows

typedef __bf16 bf16;
typedef __attribute__((ext_vector_type(16))) __bf16 v16bf;
typedef __attribute__((ext_vector_type(8)))  float  v8f;

union Frag16 { v16bf v; uint4 q[2]; };

__device__ __forceinline__ v8f zero8() {
  v8f z;
#pragma unroll
  for (int i = 0; i < 8; i++) z[i] = 0.f;
  return z;
}

// ---- DPP cross-lane reduction within each 16-lane row (no LDS traffic) ----
template <int CTRL>
__device__ __forceinline__ float dpp_movf(float x) {
  int r = __builtin_amdgcn_mov_dpp(__float_as_int(x), CTRL, 0xF, 0xF, true);
  return __int_as_float(r);
}
// row_ror:N = 0x120 + N ; rotation-based reduce gives all 16 lanes the result
__device__ __forceinline__ float row_max16(float x) {
  x = fmaxf(x, dpp_movf<0x121>(x));
  x = fmaxf(x, dpp_movf<0x122>(x));
  x = fmaxf(x, dpp_movf<0x124>(x));
  x = fmaxf(x, dpp_movf<0x128>(x));
  return x;
}
__device__ __forceinline__ float row_sum16(float x) {
  x += dpp_movf<0x121>(x);
  x += dpp_movf<0x122>(x);
  x += dpp_movf<0x124>(x);
  x += dpp_movf<0x128>(x);
  return x;
}

// ---------------------------------------------------------------- cast f32->bf16
__global__ void cast_f32_bf16(const float* __restrict__ src, bf16* __restrict__ dst, size_t n) {
  size_t i = (size_t)blockIdx.x * blockDim.x + threadIdx.x;
  size_t stride = (size_t)gridDim.x * blockDim.x;
  for (; i < n; i += stride) dst[i] = (bf16)src[i];
}

// ---------------------------------------------------------------- LayerNorm -> bf16
__global__ __launch_bounds__(256) void layernorm_bf16(
    const float* __restrict__ x, const float* __restrict__ w,
    const float* __restrict__ b, bf16* __restrict__ y) {
  int row = blockIdx.x, tid = threadIdx.x;
  const float* xr = x + (size_t)row * H_;
  __shared__ float sb[256];
  float s = 0.f;
  for (int i = tid; i < H_; i += 256) s += xr[i];
  sb[tid] = s; __syncthreads();
  for (int o = 128; o > 0; o >>= 1) { if (tid < o) sb[tid] += sb[tid + o]; __syncthreads(); }
  float mu = sb[0] * (1.f / H_); __syncthreads();
  float vv = 0.f;
  for (int i = tid; i < H_; i += 256) { float d = xr[i] - mu; vv += d * d; }
  sb[tid] = vv; __syncthreads();
  for (int o = 128; o > 0; o >>= 1) { if (tid < o) sb[tid] += sb[tid + o]; __syncthreads(); }
  float rstd = rsqrtf(sb[0] * (1.f / H_) + 1e-5f);
  bf16* yr = y + (size_t)row * H_;
  for (int i = tid; i < H_; i += 256) yr[i] = (bf16)((xr[i] - mu) * rstd * w[i] + b[i]);
}

// ---------------------------------------------------------------- GEMM: C[M,N] = A[M,K] * W[N,K]^T + bias (+epilogue)
// EPI: 0 = none, 1 = exact GELU, 2 = add residual[M,N] (fp32)
// K staged 64 per LDS stage: 16 WMMAs between barrier pairs.
template <typename OutT, int EPI>
__global__ __launch_bounds__(256) void gemm_bf16(
    const bf16* __restrict__ A, const bf16* __restrict__ W,
    const float* __restrict__ bias, const float* __restrict__ res,
    OutT* __restrict__ out, int N, int K) {
  const int LD = 72;                       // 64 halves + 8 pad (16B aligned rows: 144B)
  __shared__ bf16 At[128 * LD];
  __shared__ bf16 Wt[128 * LD];
  int tid = threadIdx.x;
  int wave = tid >> 5, lane = tid & 31, lh = lane & 15;
  bool hi = lane >= 16;
  int blockM = blockIdx.x * 128, blockN = blockIdx.y * 128;
  int wM = (wave >> 1) * 32, wN = (wave & 1) * 64;   // wave tile 32x64

  v8f acc[2][4];
#pragma unroll
  for (int fm = 0; fm < 2; fm++)
#pragma unroll
    for (int fn = 0; fn < 4; fn++) acc[fm][fn] = zero8();

  int lr = tid & 127, lc = (tid >> 7) * 32;          // cooperative tile loaders (32 halves each)
  const bf16* Ag = A + (size_t)(blockM + lr) * K + lc;
  const bf16* Wg = W + (size_t)(blockN + lr) * K + lc;
  bf16* Ats = &At[lr * LD + lc];
  bf16* Wts = &Wt[lr * LD + lc];

  for (int kk = 0; kk < K; kk += 64) {
    uint4 a0 = *(const uint4*)(Ag + kk);
    uint4 a1 = *(const uint4*)(Ag + kk + 8);
    uint4 a2 = *(const uint4*)(Ag + kk + 16);
    uint4 a3 = *(const uint4*)(Ag + kk + 24);
    uint4 w0 = *(const uint4*)(Wg + kk);
    uint4 w1 = *(const uint4*)(Wg + kk + 8);
    uint4 w2 = *(const uint4*)(Wg + kk + 16);
    uint4 w3 = *(const uint4*)(Wg + kk + 24);
    if (kk + 64 < K) {                               // prefetch next tiles into cache
      __builtin_prefetch(Ag + kk + 64, 0, 1);
      __builtin_prefetch(Wg + kk + 64, 0, 1);
    }
    __syncthreads();
    *(uint4*)(Ats)      = a0;  *(uint4*)(Ats + 8)  = a1;
    *(uint4*)(Ats + 16) = a2;  *(uint4*)(Ats + 24) = a3;
    *(uint4*)(Wts)      = w0;  *(uint4*)(Wts + 8)  = w1;
    *(uint4*)(Wts + 16) = w2;  *(uint4*)(Wts + 24) = w3;
    __syncthreads();

#pragma unroll
    for (int ks = 0; ks < 2; ks++) {                 // two K=32 substeps per stage
      int base = ks * 32;
      Frag16 af[2], bfr[4];
#pragma unroll
      for (int fm = 0; fm < 2; fm++) {               // A frag: lane<16 K{0-7,16-23}, hi K{8-15,24-31}
        const bf16* ar = &At[(wM + fm * 16 + lh) * LD + base];
        af[fm].q[0] = *(const uint4*)(ar + (hi ? 8 : 0));
        af[fm].q[1] = *(const uint4*)(ar + (hi ? 24 : 16));
      }
#pragma unroll
      for (int fn = 0; fn < 4; fn++) {               // B frag: lane<16 K{0-7,16-23}, hi K{8-15,24-31} (col = lh)
        const bf16* wr = &Wt[(wN + fn * 16 + lh) * LD + base];
        bfr[fn].q[0] = *(const uint4*)(wr + (hi ? 8 : 0));
        bfr[fn].q[1] = *(const uint4*)(wr + (hi ? 24 : 16));
      }
#pragma unroll
      for (int fm = 0; fm < 2; fm++)
#pragma unroll
        for (int fn = 0; fn < 4; fn++)
          acc[fm][fn] = __builtin_amdgcn_wmma_f32_16x16x32_bf16(
              false, af[fm].v, false, bfr[fn].v, (short)0, acc[fm][fn], false, false);
    }
  }

#pragma unroll
  for (int fm = 0; fm < 2; fm++)
#pragma unroll
    for (int fn = 0; fn < 4; fn++) {
      int ng = blockN + wN + fn * 16 + lh;
      float bn = bias[ng];
#pragma unroll
      for (int i = 0; i < 8; i++) {
        int mg = blockM + wM + fm * 16 + (hi ? 8 + i : i);
        float v = acc[fm][fn][i] + bn;
        if (EPI == 1) v = 0.5f * v * (1.f + erff(v * 0.70710678f));
        if (EPI == 2) v += res[(size_t)mg * N + ng];
        out[(size_t)mg * N + ng] = (OutT)v;
      }
    }
}

// ---------------------------------------------------------------- RoPE + head split: qkv[B,S,NH,3D] -> Q/K/V [B,NH,S,D]
__global__ __launch_bounds__(128) void rope_split(
    const bf16* __restrict__ qkv, bf16* __restrict__ Q,
    bf16* __restrict__ Kk, bf16* __restrict__ V) {
  int idx = blockIdx.x;
  int s = idx % S_;
  int h = (idx / S_) % NH_;
  int b = idx / (S_ * NH_);
  int d = threadIdx.x;
  const bf16* src = qkv + (size_t)(b * S_ + s) * (3 * H_) + h * (3 * D_);
  int j = d & 63;
  float inv = __expf((float)j * (-9.210340372f / 64.f));   // 10000^(-2j/128)
  float ang = (float)s * inv;
  float sn, c;
  __sincosf(ang, &sn, &c);
  float q = (float)src[d], k = (float)src[D_ + d], v = (float)src[2 * D_ + d];
  float qr, kr;
  if (d < 64) { qr = q * c - (float)src[d + 64] * sn;       kr = k * c - (float)src[D_ + d + 64] * sn; }
  else        { qr = q * c + (float)src[d - 64] * sn;       kr = k * c + (float)src[D_ + d - 64] * sn; }
  size_t o = ((size_t)(b * NH_ + h) * S_ + s) * D_ + d;
  Q[o] = (bf16)qr; Kk[o] = (bf16)kr; V[o] = (bf16)v;
}

// ---------------------------------------------------------------- causal flash attention, bf16 WMMA, fp32 online softmax
__global__ __launch_bounds__(256) void flash_attn(
    const bf16* __restrict__ Q, const bf16* __restrict__ Kk,
    const bf16* __restrict__ V, bf16* __restrict__ ctx) {
  const int LDK = 136, LDV = 40, LDP = 40;
  __shared__ bf16 Kt[32 * LDK];        // K tile   [32 keys][128 d]
  __shared__ bf16 Vt[128 * LDV];       // V^T tile [128 d][32 keys]
  __shared__ bf16 Pb[8 * 16 * LDP];    // per-wave P [16 q][32 keys]
  int bh = blockIdx.x;
  int q0 = blockIdx.y * 128;
  int tid = threadIdx.x, wave = tid >> 5, lane = tid & 31, lh = lane & 15;
  bool hi = lane >= 16;
  int b = bh / NH_, h = bh % NH_;
  const bf16* Qp = Q  + (size_t)bh * S_ * D_;
  const bf16* Kp = Kk + (size_t)bh * S_ * D_;
  const bf16* Vp = V  + (size_t)bh * S_ * D_;

  Frag16 qf[4];                                       // Q rows for this wave, K-dim = D in 4 chunks of 32
  {
    const bf16* qr = Qp + (size_t)(q0 + wave * 16 + lh) * D_;
#pragma unroll
    for (int dc = 0; dc < 4; dc++) {
      qf[dc].q[0] = *(const uint4*)(qr + dc * 32 + (hi ? 8 : 0));
      qf[dc].q[1] = *(const uint4*)(qr + dc * 32 + (hi ? 24 : 16));
    }
  }
  v8f o[8];
#pragma unroll
  for (int dt = 0; dt < 8; dt++) o[dt] = zero8();
  float l[8], mrow[8];
#pragma unroll
  for (int i = 0; i < 8; i++) { l[i] = 0.f; mrow[i] = -1e30f; }

  int kblocks = (q0 + 128) / 32;                      // uniform over the block -> barriers safe
  int ldr = tid & 31, c0 = (tid >> 5) * 16;
  for (int kb = 0; kb < kblocks; kb++) {
    __syncthreads();
    {   // cooperative K tile + transposed V tile
      const bf16* kr = Kp + (size_t)(kb * 32 + ldr) * D_ + c0;
      uint4 k0 = *(const uint4*)(kr);
      uint4 k1 = *(const uint4*)(kr + 8);
      *(uint4*)&Kt[ldr * LDK + c0]     = k0;
      *(uint4*)&Kt[ldr * LDK + c0 + 8] = k1;
      const bf16* vr = Vp + (size_t)(kb * 32 + ldr) * D_ + c0;
      bf16 vbuf[16];
      *(uint4*)&vbuf[0] = *(const uint4*)(vr);
      *(uint4*)&vbuf[8] = *(const uint4*)(vr + 8);
      if (kb + 1 < kblocks) {                         // prefetch next K/V tiles
        __builtin_prefetch(kr + 32 * D_, 0, 1);
        __builtin_prefetch(vr + 32 * D_, 0, 1);
      }
#pragma unroll
      for (int j = 0; j < 16; j++) Vt[(c0 + j) * LDV + ldr] = vbuf[j];
    }
    __syncthreads();

    // scores S = Q K^T (16 q x 32 keys): 2 n-tiles x 4 d-chunks
    v8f sacc[2]; sacc[0] = zero8(); sacc[1] = zero8();
#pragma unroll
    for (int dc = 0; dc < 4; dc++)
#pragma unroll
      for (int nt = 0; nt < 2; nt++) {
        Frag16 kf;
        const bf16* krow = &Kt[(nt * 16 + lh) * LDK + dc * 32];
        kf.q[0] = *(const uint4*)(krow + (hi ? 8 : 0));
        kf.q[1] = *(const uint4*)(krow + (hi ? 24 : 16));
        sacc[nt] = __builtin_amdgcn_wmma_f32_16x16x32_bf16(
            false, qf[dc].v, false, kf.v, (short)0, sacc[nt], false, false);
      }

    // online softmax in C-fragment layout (DPP row reductions, no LDS)
    float e0[8], e1[8];
#pragma unroll
    for (int i = 0; i < 8; i++) {
      int qg = q0 + wave * 16 + (hi ? 8 + i : i);
      float s0 = sacc[0][i] * 0.08838834764f; if (kb * 32 + lh      > qg) s0 = -1e9f;
      float s1 = sacc[1][i] * 0.08838834764f; if (kb * 32 + 16 + lh > qg) s1 = -1e9f;
      float tm = row_max16(fmaxf(s0, s1));
      float nm = fmaxf(mrow[i], tm);
      float alpha = __expf(mrow[i] - nm);
      mrow[i] = nm;
      e0[i] = __expf(s0 - nm); e1[i] = __expf(s1 - nm);
      float ps = row_sum16(e0[i] + e1[i]);
      l[i] = l[i] * alpha + ps;
#pragma unroll
      for (int dt = 0; dt < 8; dt++) o[dt][i] *= alpha;
    }

    // P: C-layout -> A-layout via per-wave LDS (DS ops are in-order per wave)
    bf16* pw = &Pb[wave * 16 * LDP];
#pragma unroll
    for (int i = 0; i < 8; i++) {
      int m = hi ? 8 + i : i;
      pw[m * LDP + lh]      = (bf16)e0[i];
      pw[m * LDP + 16 + lh] = (bf16)e1[i];
    }
    asm volatile("s_wait_dscnt 0" ::: "memory");
    Frag16 pf;
    {
      const bf16* pr = &pw[lh * LDP];
      pf.q[0] = *(const uint4*)(pr + (hi ? 8 : 0));
      pf.q[1] = *(const uint4*)(pr + (hi ? 24 : 16));
    }

    // O += P V : 8 d-tiles of 16
#pragma unroll
    for (int dt = 0; dt < 8; dt++) {
      Frag16 vf;
      const bf16* vrow = &Vt[(dt * 16 + lh) * LDV];
      vf.q[0] = *(const uint4*)(vrow + (hi ? 8 : 0));
      vf.q[1] = *(const uint4*)(vrow + (hi ? 24 : 16));
      o[dt] = __builtin_amdgcn_wmma_f32_16x16x32_bf16(
          false, pf.v, false, vf.v, (short)0, o[dt], false, false);
    }
  }

  // normalize and write ctx [B,S,H]
#pragma unroll
  for (int i = 0; i < 8; i++) {
    float invl = 1.f / l[i];
    int s = q0 + wave * 16 + (hi ? 8 + i : i);
    bf16* cr = ctx + ((size_t)(b * S_) + s) * H_ + h * D_;
#pragma unroll
    for (int dt = 0; dt < 8; dt++) cr[dt * 16 + lh] = (bf16)(o[dt][i] * invl);
  }
}

// ----------------------------------------------------------------
extern "C" void kernel_launch(void* const* d_in, const int* in_sizes, int n_in,
                              void* d_out, int out_size, void* d_ws, size_t ws_size,
                              hipStream_t stream) {
  (void)in_sizes; (void)n_in; (void)ws_size;
  const float* hidden_states = (const float*)d_in[0];
  const float* ln1_w  = (const float*)d_in[2];
  const float* ln1_b  = (const float*)d_in[3];
  const float* wqkv   = (const float*)d_in[4];
  const float* bqkv   = (const float*)d_in[5];
  const float* wdense = (const float*)d_in[6];
  const float* bdense = (const float*)d_in[7];
  const float* ln2_w  = (const float*)d_in[8];
  const float* ln2_b  = (const float*)d_in[9];
  const float* w1     = (const float*)d_in[10];
  const float* b1     = (const float*)d_in[11];
  const float* w2     = (const float*)d_in[12];
  const float* b2     = (const float*)d_in[13];
  float* out = (float*)d_out;

  char* p = (char*)d_ws;
  auto carve = [&](size_t bytes) -> char* {
    char* r = p; p += (bytes + 255) & ~(size_t)255; return r;
  };
  bf16*  wqkv_bf   = (bf16*)carve((size_t)3 * H_ * H_ * 2);
  bf16*  wdense_bf = (bf16*)carve((size_t)H_ * H_ * 2);
  bf16*  w1_bf     = (bf16*)carve((size_t)FF_ * H_ * 2);
  bf16*  w2_bf     = (bf16*)carve((size_t)H_ * FF_ * 2);
  bf16*  x_bf      = (bf16*)carve((size_t)MTOT * H_ * 2);
  bf16*  qkv_bf    = (bf16*)carve((size_t)MTOT * 3 * H_ * 2);
  bf16*  Qr        = (bf16*)carve((size_t)MTOT * H_ * 2);
  bf16*  Kr        = (bf16*)carve((size_t)MTOT * H_ * 2);
  bf16*  Vr        = (bf16*)carve((size_t)MTOT * H_ * 2);
  bf16*  ctx_bf    = (bf16*)carve((size_t)MTOT * H_ * 2);
  float* hid       = (float*)carve((size_t)MTOT * H_ * 4);
  bf16*  inter_bf  = (bf16*)carve((size_t)MTOT * FF_ * 2);

  cast_f32_bf16<<<2048, 256, 0, stream>>>(wqkv,   wqkv_bf,   (size_t)3 * H_ * H_);
  cast_f32_bf16<<<2048, 256, 0, stream>>>(wdense, wdense_bf, (size_t)H_ * H_);
  cast_f32_bf16<<<2048, 256, 0, stream>>>(w1,     w1_bf,     (size_t)FF_ * H_);
  cast_f32_bf16<<<2048, 256, 0, stream>>>(w2,     w2_bf,     (size_t)H_ * FF_);

  layernorm_bf16<<<MTOT, 256, 0, stream>>>(hidden_states, ln1_w, ln1_b, x_bf);

  gemm_bf16<bf16, 0><<<dim3(MTOT / 128, (3 * H_) / 128), 256, 0, stream>>>(
      x_bf, wqkv_bf, bqkv, nullptr, qkv_bf, 3 * H_, H_);

  rope_split<<<B_ * NH_ * S_, 128, 0, stream>>>(qkv_bf, Qr, Kr, Vr);

  flash_attn<<<dim3(B_ * NH_, S_ / 128), 256, 0, stream>>>(Qr, Kr, Vr, ctx_bf);

  gemm_bf16<float, 2><<<dim3(MTOT / 128, H_ / 128), 256, 0, stream>>>(
      ctx_bf, wdense_bf, bdense, hidden_states, hid, H_, H_);

  layernorm_bf16<<<MTOT, 256, 0, stream>>>(hid, ln2_w, ln2_b, x_bf);

  gemm_bf16<bf16, 1><<<dim3(MTOT / 128, FF_ / 128), 256, 0, stream>>>(
      x_bf, w1_bf, b1, nullptr, inter_bf, FF_, H_);

  gemm_bf16<float, 2><<<dim3(MTOT / 128, H_ / 128), 256, 0, stream>>>(
      inter_bf, w2_bf, b2, hid, out, H_, FF_);

  // second tuple element: attention_mask (all-False bools) — zero the tail
  size_t bsh = (size_t)MTOT * H_;
  if ((size_t)out_size > bsh)
    hipMemsetAsync((char*)d_out + bsh * sizeof(float), 0, (size_t)out_size - bsh, stream);
}
